// MultiheadAttention_71064528879633
// MI455X (gfx1250) — matrix operand order repaired and checked by reading, not compiled, not run on hardware
//
#include <hip/hip_runtime.h>

typedef __attribute__((ext_vector_type(16))) __bf16 v16bf;
typedef __attribute__((ext_vector_type(8)))  __bf16 v8bf;
typedef __attribute__((ext_vector_type(8)))  float  v8f;

// ---------- helpers ----------
static __device__ __forceinline__ __bf16 tobf(float f) {
  unsigned u = __builtin_bit_cast(unsigned, f);
  unsigned r = u + 0x7FFFu + ((u >> 16) & 1u);   // round-to-nearest-even
  unsigned short h = (unsigned short)(r >> 16);
  return __builtin_bit_cast(__bf16, h);
}

// Build a v16bf fragment from two contiguous 8-element (16B) chunks.
static __device__ __forceinline__ v16bf frag2(const __bf16* p0, const __bf16* p1) {
  v8bf lo = *(const v8bf*)p0;
  v8bf hi = *(const v8bf*)p1;
  v16bf r;
#pragma unroll
  for (int i = 0; i < 8; ++i) { r[i] = lo[i]; r[i + 8] = hi[i]; }
  return r;
}

// ---------- segment bookkeeping ----------
__global__ void k_zero(int* counts) { if (threadIdx.x < 16) counts[threadIdx.x] = 0; }

__global__ void k_count(const int* __restrict__ ids, int* __restrict__ counts, int N) {
  int i = blockIdx.x * blockDim.x + threadIdx.x;
  if (i < N) atomicAdd(&counts[ids[i]], 1);
}

__global__ void k_scan(const int* __restrict__ counts, int* __restrict__ offs) {
  int a = 0; offs[0] = 0;
  for (int b = 0; b < 16; ++b) { a += counts[b]; offs[b + 1] = a; }
}

// ---------- conversions / weight packing ----------
__global__ void k_cvt_x(const float* __restrict__ x, __bf16* __restrict__ xb, int n) {
  int i = blockIdx.x * blockDim.x + threadIdx.x;
  if (i < n) xb[i] = tobf(x[i]);
}

// Pack w[Nout,Kdim] (row-major, fp32) into WMMA B-fragment order:
// wp[((kt*NT + nt)*32 + lane)*16 + e] = bf16(w[n*Kdim + k]),
//   k = kt*32 + (lane>>4)*16 + e,  n = nt*16 + (lane&15)
__global__ void k_pack_w(const float* __restrict__ w, __bf16* __restrict__ wp,
                         int Kdim, int Nout) {
  int i = blockIdx.x * blockDim.x + threadIdx.x;
  if (i >= Kdim * Nout) return;
  int e = i & 15;
  int lane = (i >> 4) & 31;
  int t = i >> 9;                 // kt*NT + nt
  int NT = Nout >> 4;
  int nt = t % NT, kt = t / NT;
  int k = kt * 32 + ((lane >> 4) * 16) + e;
  int n = nt * 16 + (lane & 15);
  wp[i] = tobf(w[(size_t)n * Kdim + k]);
}

// ---------- QKV projection: x[N,256] @ qkv_w^T + b, 16x64 tile per wave ----------
// Writes Q/K row-major [H][Npad][32] and V transposed [H][32][Npad], all bf16.
__global__ void __launch_bounds__(32)
k_gemm_qkv(const __bf16* __restrict__ xb, const __bf16* __restrict__ wp,
           const float* __restrict__ bias,
           __bf16* __restrict__ Q, __bf16* __restrict__ K, __bf16* __restrict__ Vt,
           int N, int Npad) {
  const int mt = blockIdx.x, nt4 = blockIdx.y * 4;   // 4 N-tiles per wave
  const int lane = threadIdx.x, m = lane & 15, hi = lane >> 4;
  const int NT = 48;                                 // 768/16
  const __bf16* arow = xb + (size_t)(mt * 16 + m) * 256;
  v8f c0 = {}, c1 = {}, c2 = {}, c3 = {};
#pragma unroll
  for (int kt = 0; kt < 8; ++kt) {                   // K = 256, 8 steps of 32
    v16bf a = frag2(arow + kt * 32 + hi * 16, arow + kt * 32 + hi * 16 + 8);
    const __bf16* bp = wp + ((size_t)(kt * NT + nt4) * 32 + lane) * 16;
    v16bf b0 = frag2(bp,            bp + 8);
    v16bf b1 = frag2(bp + 32 * 16,  bp + 32 * 16 + 8);
    v16bf b2 = frag2(bp + 64 * 16,  bp + 64 * 16 + 8);
    v16bf b3 = frag2(bp + 96 * 16,  bp + 96 * 16 + 8);
    c0 = __builtin_amdgcn_wmma_f32_16x16x32_bf16(false, a, false, b0, (short)0, c0, false, false);
    c1 = __builtin_amdgcn_wmma_f32_16x16x32_bf16(false, a, false, b1, (short)0, c1, false, false);
    c2 = __builtin_amdgcn_wmma_f32_16x16x32_bf16(false, a, false, b2, (short)0, c2, false, false);
    c3 = __builtin_amdgcn_wmma_f32_16x16x32_bf16(false, a, false, b3, (short)0, c3, false, false);
  }
  v8f cs[4] = {c0, c1, c2, c3};
#pragma unroll
  for (int nn = 0; nn < 4; ++nn) {
    const int n = (nt4 + nn) * 16 + m;
    const float bn = bias[n];
    const int h = n / 96, r96 = n % 96;              // per head: [q(32) | k(32) | v(32)]
#pragma unroll
    for (int r = 0; r < 8; ++r) {
      int tok = mt * 16 + hi * 8 + r;
      if (tok >= N) continue;
      __bf16 v = tobf(cs[nn][r] + bn);
      if (r96 < 32)      Q[((size_t)h * Npad + tok) * 32 + r96] = v;
      else if (r96 < 64) K[((size_t)h * Npad + tok) * 32 + (r96 - 32)] = v;
      else               Vt[(size_t)h * 32 * Npad + (size_t)(r96 - 64) * Npad + tok] = v;
    }
  }
}

// ---------- flash attention per (batch, head, 16-query tile), head_dim = 32 ----------
__global__ void __launch_bounds__(32)
k_attn(const __bf16* __restrict__ Q, const __bf16* __restrict__ K,
       const __bf16* __restrict__ Vt, __bf16* __restrict__ att,
       const int* __restrict__ offs, int Npad) {
  __shared__ __bf16 pst[16 * 32];
  const int b = blockIdx.z, h = blockIdx.y, qt = blockIdx.x;
  const int s0 = offs[b], S = offs[b + 1] - s0;
  if (qt * 16 >= S) return;                  // wave-uniform exit
  const int lane = threadIdx.x, m = lane & 15, hi = lane >> 4;
  const float SC = 0.17677669529663687f;     // 1/sqrt(32)

  const __bf16* qrow = Q + ((size_t)h * Npad + s0 + qt * 16 + m) * 32;
  const v16bf aq = frag2(qrow + hi * 16, qrow + hi * 16 + 8);

  float mrow[8], lrow[8];
#pragma unroll
  for (int r = 0; r < 8; ++r) { mrow[r] = -3.4e38f; lrow[r] = 0.f; }
  v8f acc0 = {}, acc1 = {};

  const __bf16* kbase = K + ((size_t)h * Npad + s0) * 32;
  const __bf16* vbase = Vt + (size_t)h * 32 * Npad + s0;

  for (int kb = 0; kb < S; kb += 32) {
    // logits: two 16x16 tiles, K-dim = head_dim = 32 (one WMMA each)
    const __bf16* k0 = kbase + (size_t)(kb + m) * 32;
    const __bf16* k1 = kbase + (size_t)(kb + 16 + m) * 32;
    v16bf bk0 = frag2(k0 + hi * 16, k0 + hi * 16 + 8);
    v16bf bk1 = frag2(k1 + hi * 16, k1 + hi * 16 + 8);
    v8f c0 = {}, c1 = {};
    c0 = __builtin_amdgcn_wmma_f32_16x16x32_bf16(false, aq, false, bk0, (short)0, c0, false, false);
    c1 = __builtin_amdgcn_wmma_f32_16x16x32_bf16(false, aq, false, bk1, (short)0, c1, false, false);
    const bool ok0 = (kb + m) < S, ok1 = (kb + 16 + m) < S;
#pragma unroll
    for (int r = 0; r < 8; ++r) {
      c0[r] = ok0 ? c0[r] * SC : -3.4e38f;
      c1[r] = ok1 ? c1[r] * SC : -3.4e38f;
    }
    // online softmax: stats per row M = hi*8+r live across 16-lane half
#pragma unroll
    for (int r = 0; r < 8; ++r) {
      float v = fmaxf(c0[r], c1[r]);
#pragma unroll
      for (int s = 1; s < 16; s <<= 1) v = fmaxf(v, __shfl_xor(v, s, 32));
      float nm = fmaxf(mrow[r], v);
      float alpha = __expf(mrow[r] - nm);
      float p0 = __expf(c0[r] - nm);
      float p1 = __expf(c1[r] - nm);
      float ssum = p0 + p1;
#pragma unroll
      for (int s = 1; s < 16; s <<= 1) ssum += __shfl_xor(ssum, s, 32);
      lrow[r] = lrow[r] * alpha + ssum;
      mrow[r] = nm;
      acc0[r] *= alpha; acc1[r] *= alpha;
      pst[(hi * 8 + r) * 32 + m]      = tobf(p0);   // P row-major [q,key] for transpose
      pst[(hi * 8 + r) * 32 + 16 + m] = tobf(p1);
    }
    __syncthreads();
    const __bf16* prow = pst + m * 32;
    v16bf ap = frag2(prow + hi * 16, prow + hi * 16 + 8);   // A-frag of P (16q x 32keys)
    __syncthreads();
    // P @ V: V transposed rows are contiguous in key -> pure b128 loads
    const __bf16* v0 = vbase + (size_t)m * Npad + kb + hi * 16;
    const __bf16* v1 = vbase + (size_t)(16 + m) * Npad + kb + hi * 16;
    v16bf bv0 = frag2(v0, v0 + 8);
    v16bf bv1 = frag2(v1, v1 + 8);
    acc0 = __builtin_amdgcn_wmma_f32_16x16x32_bf16(false, ap, false, bv0, (short)0, acc0, false, false);
    acc1 = __builtin_amdgcn_wmma_f32_16x16x32_bf16(false, ap, false, bv1, (short)0, acc1, false, false);
  }

#pragma unroll
  for (int r = 0; r < 8; ++r) {
    int row = qt * 16 + hi * 8 + r;
    if (row >= S) continue;
    float inv = 1.0f / lrow[r];
    size_t tok = (size_t)(s0 + row);
    att[tok * 256 + h * 32 + m]      = tobf(acc0[r] * inv);
    att[tok * 256 + h * 32 + 16 + m] = tobf(acc1[r] * inv);
  }
}

// ---------- output projection: att[N,256] @ o_w^T + o_b, 16x64 tile per wave ----------
__global__ void __launch_bounds__(32)
k_gemm_out(const __bf16* __restrict__ ab, const __bf16* __restrict__ wp,
           const float* __restrict__ bias, float* __restrict__ out, int N) {
  const int mt = blockIdx.x, nt4 = blockIdx.y * 4;
  const int lane = threadIdx.x, m = lane & 15, hi = lane >> 4;
  const int NT = 16;                         // 256/16
  const __bf16* arow = ab + (size_t)(mt * 16 + m) * 256;
  v8f c0 = {}, c1 = {}, c2 = {}, c3 = {};
#pragma unroll
  for (int kt = 0; kt < 8; ++kt) {
    v16bf a = frag2(arow + kt * 32 + hi * 16, arow + kt * 32 + hi * 16 + 8);
    const __bf16* bp = wp + ((size_t)(kt * NT + nt4) * 32 + lane) * 16;
    v16bf b0 = frag2(bp,            bp + 8);
    v16bf b1 = frag2(bp + 32 * 16,  bp + 32 * 16 + 8);
    v16bf b2 = frag2(bp + 64 * 16,  bp + 64 * 16 + 8);
    v16bf b3 = frag2(bp + 96 * 16,  bp + 96 * 16 + 8);
    c0 = __builtin_amdgcn_wmma_f32_16x16x32_bf16(false, a, false, b0, (short)0, c0, false, false);
    c1 = __builtin_amdgcn_wmma_f32_16x16x32_bf16(false, a, false, b1, (short)0, c1, false, false);
    c2 = __builtin_amdgcn_wmma_f32_16x16x32_bf16(false, a, false, b2, (short)0, c2, false, false);
    c3 = __builtin_amdgcn_wmma_f32_16x16x32_bf16(false, a, false, b3, (short)0, c3, false, false);
  }
  v8f cs[4] = {c0, c1, c2, c3};
#pragma unroll
  for (int nn = 0; nn < 4; ++nn) {
    const int n = (nt4 + nn) * 16 + m;
    const float bn = bias[n];
#pragma unroll
    for (int r = 0; r < 8; ++r) {
      int tok = mt * 16 + hi * 8 + r;
      if (tok < N) out[(size_t)tok * 256 + n] = cs[nn][r] + bn;
    }
  }
}

// ---------- launch ----------
extern "C" void kernel_launch(void* const* d_in, const int* in_sizes, int n_in,
                              void* d_out, int out_size, void* d_ws, size_t ws_size,
                              hipStream_t stream) {
  const float* x     = (const float*)d_in[0];
  const int*   ids   = (const int*)d_in[1];
  const float* qkv_w = (const float*)d_in[2];
  const float* qkv_b = (const float*)d_in[3];
  const float* o_w   = (const float*)d_in[4];
  const float* o_b   = (const float*)d_in[5];
  const int N    = in_sizes[1];              // token count
  const int Mt   = (N + 15) / 16;
  const int Nrnd = Mt * 16;
  const int Npad = Nrnd + 64;                // slack so tail fragment loads never fault

  char* p = (char*)d_ws;
  auto alloc = [&](size_t bytes) { char* r = p; p += (bytes + 255) & ~(size_t)255; return r; };
  int*    offs   = (int*)alloc(64 * sizeof(int));
  int*    counts = offs + 32;
  __bf16* xb  = (__bf16*)alloc((size_t)Nrnd * 256 * 2);
  __bf16* Qb  = (__bf16*)alloc((size_t)8 * Npad * 32 * 2);
  __bf16* Kb  = (__bf16*)alloc((size_t)8 * Npad * 32 * 2);
  __bf16* Vt  = (__bf16*)alloc((size_t)8 * 32 * Npad * 2);
  __bf16* att = (__bf16*)alloc((size_t)Nrnd * 256 * 2);
  __bf16* wq  = (__bf16*)alloc((size_t)768 * 256 * 2);
  __bf16* wo  = (__bf16*)alloc((size_t)256 * 256 * 2);

  k_zero <<<1, 32, 0, stream>>>(counts);
  k_count<<<(N + 255) / 256, 256, 0, stream>>>(ids, counts, N);
  k_scan <<<1, 1, 0, stream>>>(counts, offs);
  k_cvt_x<<<(N * 256 + 255) / 256, 256, 0, stream>>>(x, xb, N * 256);
  k_pack_w<<<(768 * 256 + 255) / 256, 256, 0, stream>>>(qkv_w, wq, 256, 768);
  k_pack_w<<<(256 * 256 + 255) / 256, 256, 0, stream>>>(o_w, wo, 256, 256);

  k_gemm_qkv<<<dim3(Mt, 12), 32, 0, stream>>>(xb, wq, qkv_b, Qb, Kb, Vt, N, Npad);
  k_attn    <<<dim3(Mt, 8, 16), 32, 0, stream>>>(Qb, Kb, Vt, att, offs, Npad);
  k_gemm_out<<<dim3(Mt, 4), 32, 0, stream>>>(att, wo, o_b, (float*)d_out, N);
}
